// SmartDoorClassifierv1_39960375722575
// MI455X (gfx1250) — matrix operand order repaired and checked by reading, not compiled, not run on hardware
//
#include <hip/hip_runtime.h>

// SNN with tau_mem=0.005 => alpha = exp(-200) == 0.0f exactly in fp32:
// LIF(spiking) is elementwise floor(max(v,0)); the readout LIF is identity.
// Net = quantized CNN: conv1(2x2,s2)+Q+pool -> conv2(3x3,p1)+Q+pool ->
// conv3(3x3,p1)+Q+pool -> FC(512->2). Spike counts are small non-negative
// ints (exact in f16) -> conv2/conv3/FC run on v_wmma_f32_16x16x32_f16 with
// f32 accumulation.
//
// Data layout is chosen for the WMMA fragment geometry: activations are NHWC
// f16 and the im2col K-order is k = (ky*3+kx)*8 + c, so every 8-element
// half-fragment of B is one contiguous global_load_b128 (one 3x3 tap, all 8
// channels), and the K>=72 zero padding falls on whole halves (no loads).
// Boundary handling = per-tap v_cndmask masking, never saveexec divergence.

typedef __attribute__((ext_vector_type(16))) _Float16 v16h;
typedef __attribute__((ext_vector_type(8)))  _Float16 v8h;
typedef __attribute__((ext_vector_type(8)))  float    v8f;
typedef __attribute__((ext_vector_type(4)))  int      v4i;

#define N_IMG 1024

union V16 {
  v16h v;
  v8h  h[2];
};
union H8 {
  v8h h;
  v4i q;
};

__device__ __forceinline__ float spikeq(float v) {
  return floorf(fmaxf(v, 0.0f));  // MultiSpike, thr = 1
}

// ---------------- one-shot weight packing into f16 GEMM layout --------------
// Wp2/Wp3: 16 rows (out ch; rows 8..15 zero) x 96 K, K = (ky*3+kx)*8 + c,
//          K>=72 zero.  Wfcp: 16 rows (neurons; rows 2..15 zero) x 512 K,
//          K permuted to the NHWC flatten order: k_mem=(y*8+x)*8+c -> c*64+p.
__global__ void pack_weights(const float* __restrict__ w2,
                             const float* __restrict__ w3,
                             const float* __restrict__ wfc,
                             _Float16* __restrict__ Wp2,
                             _Float16* __restrict__ Wp3,
                             _Float16* __restrict__ Wfcp) {
  int id = blockIdx.x * blockDim.x + threadIdx.x;
  if (id < 3072) {
    int t = (id < 1536) ? id : id - 1536;
    int row = t / 96, k = t % 96;
    int g = k >> 3, c = k & 7, ky = g / 3, kx = g - 3 * ky;
    const float* w = (id < 1536) ? w2 : w3;
    _Float16* Wp  = (id < 1536) ? Wp2 : Wp3;
    Wp[t] = (row < 8 && k < 72) ? (_Float16)w[row * 72 + c * 9 + ky * 3 + kx]
                                : (_Float16)0.0f;
  } else if (id < 3072 + 8192) {
    int t = id - 3072, row = t >> 9, kmem = t & 511;
    int c = kmem & 7, p = kmem >> 3;              // NHWC -> NCHW feature
    Wfcp[t] = (row < 2) ? (_Float16)wfc[row * 512 + c * 64 + p]
                        : (_Float16)0.0f;
  }
}

// ---------------- Stage 1: conv1 (2x2, s2) + quantize + maxpool2 ------------
// x:(1024,2,128,128)f32 -> act1 NHWC:(1024,32,32,8)f16. Per-thread footprint
// is 2ch x 4 rows x 4 consecutive floats -> 8 coalesced float4 loads; all 8
// output channels packed into one b128 store.
__global__ void conv1_spike_pool(const float* __restrict__ x,
                                 const float* __restrict__ w1,
                                 _Float16* __restrict__ act1) {
  int id = blockIdx.x * blockDim.x + threadIdx.x;  // 1024*32*32
  int ox = id & 31, oy = (id >> 5) & 31, n = id >> 10;
  if (n >= N_IMG) return;

  float4 f[2][4];
#pragma unroll
  for (int ci = 0; ci < 2; ++ci)
#pragma unroll
    for (int r = 0; r < 4; ++r)
      f[ci][r] = *(const float4*)(x + ((size_t)(n * 2 + ci) * 128 + 4 * oy + r) * 128 + 4 * ox);

  v8h res;
#pragma unroll
  for (int co = 0; co < 8; ++co) {
    float wv[8];
#pragma unroll
    for (int k = 0; k < 8; ++k) wv[k] = w1[co * 8 + k];  // uniform -> SMEM
    float mx = 0.0f;  // spikes >= 0, so 0-init == -inf init
#pragma unroll
    for (int py = 0; py < 2; ++py)
#pragma unroll
      for (int px = 0; px < 2; ++px) {
        float v = 0.0f;
#pragma unroll
        for (int ci = 0; ci < 2; ++ci)
#pragma unroll
          for (int ky = 0; ky < 2; ++ky)
#pragma unroll
            for (int kx = 0; kx < 2; ++kx)
              v = fmaf(((const float*)&f[ci][2 * py + ky])[2 * px + kx],
                       wv[ci * 4 + ky * 2 + kx], v);
        mx = fmaxf(mx, spikeq(v));
      }
    res[co] = (_Float16)mx;
  }
  *(v8h*)(act1 + (((size_t)n * 32 + oy) * 32 + ox) * 8) = res;
}

// -------- Stages 2/3: 3x3 pad1 conv (8->8 ch) as WMMA implicit GEMM ---------
// One wave: 16-pixel tile x 2 adjacent rows (6 wmma, K=72 padded to 96),
// quantize, 2x2 maxpool (cross-lane shfl_xor), packed b128 NHWC store.
// B half-fragments (K spans of 8) == single (ky,kx) taps == one b128 load,
// lane-half variant chosen by one v_cndmask on a constant offset.
template <int HW>
__global__ void conv3x3_spike_pool_wmma(const _Float16* __restrict__ actIn,
                                        const _Float16* __restrict__ Wp,
                                        _Float16* __restrict__ actOut) {
  const int TPI = (HW / 2) * (HW / 16);  // wave tiles per image
  int wave = blockIdx.x * blockDim.y + threadIdx.y;
  int lane = threadIdx.x;
  int n = wave / TPI;
  if (n >= N_IMG) return;  // wave-uniform: EXEC stays full for WMMA
  int rem = wave % TPI;
  int y2 = rem / (HW / 16);
  int x0 = (rem - y2 * (HW / 16)) * 16;

  int  col = lane & 15;
  bool hi  = (lane & 16) != 0;
  int  lo  = hi ? 8 : 0;

  // A fragments: contiguous b128 loads from packed weights (zero-padded).
  V16 a[3];
#pragma unroll
  for (int cb = 0; cb < 3; ++cb) {
    a[cb].h[0] = *(const v8h*)(Wp + col * 96 + cb * 32 + 2 * lo);
    a[cb].h[1] = *(const v8h*)(Wp + col * 96 + cb * 32 + 2 * lo + 8);
  }

  int  colX = x0 + col;
  bool colOK0 = colX >= 1;
  bool colOK2 = colX <= HW - 2;
  const _Float16* base = actIn + (size_t)n * HW * HW * 8;
  const v4i zero = {0, 0, 0, 0};

#define GOFF(ky, kx) (((ky) * HW + (kx)) * 8)  // tap offset in f16 elements

  v8f d[2];
#pragma unroll
  for (int ry = 0; ry < 2; ++ry) {
    int  rowY = 2 * y2 + ry;
    bool rowOK0 = rowY >= 1;
    bool rowOK2 = rowY <= HW - 2;
    bool ok[3][3];
    ok[0][0] = rowOK0 && colOK0; ok[0][1] = rowOK0; ok[0][2] = rowOK0 && colOK2;
    ok[1][0] = colOK0;           ok[1][1] = true;   ok[1][2] = colOK2;
    ok[2][0] = rowOK2 && colOK0; ok[2][1] = rowOK2; ok[2][2] = rowOK2 && colOK2;
    // tap (ky=0,kx=0) base address for this lane's pixel column
    const _Float16* rb = base + ((size_t)(rowY - 1) * HW + (colX - 1)) * 8;

    V16 b[3];
    // half-fragment -> tap map (lane<16 | lane>=16):
    //  b0.h0: (0,0)|(0,2)   b0.h1: (0,1)|(1,0)
    //  b1.h0: (1,1)|(2,0)   b1.h1: (1,2)|(2,1)
    //  b2.h0: (2,2)|zero    b2.h1: zero |zero
    {
      H8 t;
      t.q = *(const v4i*)(rb + (hi ? GOFF(0, 2) : GOFF(0, 0)));
      t.q = (hi ? ok[0][2] : ok[0][0]) ? t.q : zero;
      b[0].h[0] = t.h;
      t.q = *(const v4i*)(rb + (hi ? GOFF(1, 0) : GOFF(0, 1)));
      t.q = (hi ? ok[1][0] : ok[0][1]) ? t.q : zero;
      b[0].h[1] = t.h;
      t.q = *(const v4i*)(rb + (hi ? GOFF(2, 0) : GOFF(1, 1)));
      t.q = (hi ? ok[2][0] : ok[1][1]) ? t.q : zero;
      b[1].h[0] = t.h;
      t.q = *(const v4i*)(rb + (hi ? GOFF(2, 1) : GOFF(1, 2)));
      t.q = (hi ? ok[2][1] : ok[1][2]) ? t.q : zero;
      b[1].h[1] = t.h;
      t.q = *(const v4i*)(rb + GOFF(2, 2));      // hi lanes: K padding -> 0
      t.q = (hi ? false : ok[2][2]) ? t.q : zero;
      b[2].h[0] = t.h;
      H8 z;
      z.q = zero;
      b[2].h[1] = z.h;                           // K=72..95 padding
    }

    v8f c = {};
    c = __builtin_amdgcn_wmma_f32_16x16x32_f16(false, a[0].v, false, b[0].v,
                                               (short)0, c, false, false);
    c = __builtin_amdgcn_wmma_f32_16x16x32_f16(false, a[1].v, false, b[1].v,
                                               (short)0, c, false, false);
    c = __builtin_amdgcn_wmma_f32_16x16x32_f16(false, a[2].v, false, b[2].v,
                                               (short)0, c, false, false);
    d[ry] = c;
  }
#undef GOFF

  // Quantize + 2x2 pool. Valid data: lanes 0..15, D rows r = out channel.
  v8h pv;
#pragma unroll
  for (int r = 0; r < 8; ++r) {
    float m = fmaxf(spikeq(d[0][r]), spikeq(d[1][r]));  // pool over y
    m = fmaxf(m, __shfl_xor(m, 1, 32));                 // pool over x
    pv[r] = (_Float16)m;
  }
  if (lane < 16 && (lane & 1) == 0) {
    int oxo = (x0 + lane) >> 1;
    *(v8h*)(actOut + (((size_t)n * (HW / 2) + y2) * (HW / 2) + oxo) * 8) = pv;
  }
}

// ---------------- Stage 4: FC 512->2 as WMMA, identity readout --------------
// One wave per 16 images; all fragment feeds are contiguous b128 loads;
// 16 fully-unrolled wmma per wave. Weight rows already match NHWC flatten.
__global__ void fc_wmma(const _Float16* __restrict__ act3,
                        const _Float16* __restrict__ Wfcp,
                        float* __restrict__ out) {
  int wave = blockIdx.x * blockDim.y + threadIdx.y;
  int lane = threadIdx.x;
  int base = wave * 16;
  if (base >= N_IMG) return;
  int m  = lane & 15;
  int lo = (lane & 16) ? 8 : 0;
  int bo = (lane & 16) ? 16 : 0;

  const _Float16* arow = act3 + (size_t)(base + m) * 512;
  const _Float16* brow = Wfcp + (size_t)m * 512;  // rows 2..15 are zeros

  v8f c = {};
#pragma unroll
  for (int kb = 0; kb < 16; ++kb) {
    V16 a, b;
    a.h[0] = *(const v8h*)(arow + kb * 32 + bo);
    a.h[1] = *(const v8h*)(arow + kb * 32 + bo + 8);
    b.h[0] = *(const v8h*)(brow + kb * 32 + bo);
    b.h[1] = *(const v8h*)(brow + kb * 32 + bo + 8);
    c = __builtin_amdgcn_wmma_f32_16x16x32_f16(false, a.v, false, b.v,
                                               (short)0, c, false, false);
  }
  if (m < 2) {
    int rowoff = (lane & 16) ? 8 : 0;  // lanes 16..31 carry D rows M=r+8
#pragma unroll
    for (int r = 0; r < 8; ++r) out[(base + rowoff + r) * 2 + m] = c[r];
  }
}

extern "C" void kernel_launch(void* const* d_in, const int* in_sizes, int n_in,
                              void* d_out, int out_size, void* d_ws,
                              size_t ws_size, hipStream_t stream) {
  const float* x   = (const float*)d_in[0];  // (1024,2,128,128)
  const float* w1  = (const float*)d_in[1];  // (8,2,2,2)
  const float* w2  = (const float*)d_in[2];  // (8,8,3,3)
  const float* w3  = (const float*)d_in[3];  // (8,8,3,3)
  const float* wfc = (const float*)d_in[4];  // (2,512)
  float* out = (float*)d_out;                // (1024,2) f32

  char* ws = (char*)d_ws;
  _Float16* Wp2  = (_Float16*)(ws);                      //  3 KiB (16x96)
  _Float16* Wp3  = (_Float16*)(ws + 4 * 1024);           //  3 KiB
  _Float16* Wfcp = (_Float16*)(ws + 8 * 1024);           // 16 KiB (16x512)
  _Float16* act1 = (_Float16*)(ws + 64 * 1024);                            // 16 MiB NHWC
  _Float16* act2 = (_Float16*)(ws + 64 * 1024 + (size_t)16 * 1024 * 1024); //  4 MiB NHWC
  _Float16* act3 = (_Float16*)(ws + 64 * 1024 + (size_t)20 * 1024 * 1024); //  1 MiB NHWC

  pack_weights<<<44, 256, 0, stream>>>(w2, w3, wfc, Wp2, Wp3, Wfcp);
  conv1_spike_pool<<<4096, 256, 0, stream>>>(x, w1, act1);

  dim3 blk(32, 4);  // 4 waves per block (wave32)
  conv3x3_spike_pool_wmma<32><<<8192, blk, 0, stream>>>(act1, Wp2, act2);  // 32768 waves
  conv3x3_spike_pool_wmma<16><<<2048, blk, 0, stream>>>(act2, Wp3, act3);  //  8192 waves
  fc_wmma<<<16, blk, 0, stream>>>(act3, Wfcp, out);                        //    64 waves
}
